// StrategySelector_91070486545012
// MI455X (gfx1250) — compile-verified, matrix operand order checked by reading, not run on hardware
//
#include <hip/hip_runtime.h>
#include <stdint.h>

// ---------------------------------------------------------------------------
// Types for CDNA5 WMMA
// ---------------------------------------------------------------------------
typedef __attribute__((ext_vector_type(16))) _Float16 v16h;
typedef __attribute__((ext_vector_type(4)))  _Float16 v4h;
typedef __attribute__((ext_vector_type(8)))  float    v8f;

#define D_IN   4096
#define HDIM   1024
#define NROLE  8
#define NEDGE  4096
#define NLOOP  1024
#define EPS_F  1e-5f

__device__ __forceinline__ v8f wmma_f16(v16h a, v16h b, v8f c) {
    // D = A(16x32 f16) * B(32x16 f16) + C(16x16 f32)
    return __builtin_amdgcn_wmma_f32_16x16x32_f16(
        /*neg_a=*/false, a, /*neg_b=*/false, b,
        /*c_mod=*/(short)0, c, /*reuse_a=*/false, /*reuse_b=*/false);
}

__device__ __forceinline__ v8f v8f_zero() {
    v8f z;
#pragma unroll
    for (int e = 0; e < 8; ++e) z[e] = 0.0f;
    return z;
}

// Per-lane fragment from two groups of 8 consecutive fp32 (converted).
__device__ __forceinline__ v16h frag_from_f32(const float* __restrict__ p0,
                                              const float* __restrict__ p1) {
    v16h f;
#pragma unroll
    for (int t = 0; t < 8; ++t) f[t] = (_Float16)p0[t];
#pragma unroll
    for (int t = 0; t < 8; ++t) f[8 + t] = (_Float16)p1[t];
    return f;
}

// Per-lane fragment from two groups of 8 consecutive f16 (16B loads each).
__device__ __forceinline__ v16h frag_from_f16(const _Float16* __restrict__ p0,
                                              const _Float16* __restrict__ p1) {
    v16h f;
#pragma unroll
    for (int t = 0; t < 8; ++t) f[t] = p0[t];
#pragma unroll
    for (int t = 0; t < 8; ++t) f[8 + t] = p1[t];
    return f;
}

// ---------------------------------------------------------------------------
// Kernel: C16[m,:] = l2norm_row( A_f32[M,K] * B[1024,K]^T + bias )  (f16 out)
// B is f16 when B_IS_F16 (pre-converted weights), else f32 (cvt in-loop).
// Block: 256 thr (8 waves). Block tile = 32 M x 1024 N. Wave tile = 32M x 128N.
// A tile staged through LDS as f16 (converted once by the stagers).
// grid.x = M/32. K multiple of 32.
// ---------------------------------------------------------------------------
template <bool B_IS_F16>
__global__ __launch_bounds__(256) void k_gemm_norm(
    const float* __restrict__ A, const void* __restrict__ Bv,
    const float* __restrict__ bias, _Float16* __restrict__ out16, int K)
{
    __shared__ _Float16 Ah[32][40];   // 32 rows x 32 k (f16), stride 40 (80B, 16B-aligned)
    __shared__ float rowsq[32];

    const _Float16* __restrict__ Bh = (const _Float16*)Bv;
    const float*    __restrict__ Bf = (const float*)Bv;

    const int tid  = threadIdx.x;
    const int wave = tid >> 5;
    const int lane = tid & 31;
    const int half = lane >> 4;   // 0/1
    const int r    = lane & 15;
    const int m0   = blockIdx.x * 32;
    const int n0   = wave * 128;

    v8f acc[2][8];
#pragma unroll
    for (int mt = 0; mt < 2; ++mt)
#pragma unroll
        for (int nt = 0; nt < 8; ++nt) acc[mt][nt] = v8f_zero();

    const int srow = tid >> 3;          // 0..31
    const int sc4  = (tid & 7) << 2;    // 0..28 step 4

    for (int kk = 0; kk < K; kk += 32) {
        // ---- stage A tile (32x32) through LDS, converting f32 -> f16 ----
        {
            const float* src = A + (size_t)(m0 + srow) * K + kk + sc4;
            float4 v = *(const float4*)src;
            v4h h;
            h[0] = (_Float16)v.x; h[1] = (_Float16)v.y;
            h[2] = (_Float16)v.z; h[3] = (_Float16)v.w;
            *(v4h*)&Ah[srow][sc4] = h;
            if (kk + 32 < K)
                __builtin_prefetch(src + 32, 0, 1);   // gfx1250 global_prefetch_b8
        }
        __syncthreads();

        v16h afr[2];
#pragma unroll
        for (int mt = 0; mt < 2; ++mt) {
            const _Float16* row = &Ah[mt * 16 + r][0];
            afr[mt] = frag_from_f16(row + half * 8, row + 16 + half * 8);
        }
#pragma unroll
        for (int nt = 0; nt < 8; ++nt) {
            v16h bfr;
            if (B_IS_F16) {
                const _Float16* brow = Bh + (size_t)(n0 + nt * 16 + r) * K + kk;
                bfr = frag_from_f16(brow + half * 8, brow + 16 + half * 8);
            } else {
                const float* brow = Bf + (size_t)(n0 + nt * 16 + r) * K + kk;
                bfr = frag_from_f32(brow + half * 8, brow + 16 + half * 8);
            }
            acc[0][nt] = wmma_f16(afr[0], bfr, acc[0][nt]);
            acc[1][nt] = wmma_f16(afr[1], bfr, acc[1][nt]);
        }
        __syncthreads();
    }

    // ---- epilogue: +bias, fused row L2-norm over full N=1024, f16 store ----
    if (tid < 32) rowsq[tid] = 0.0f;
    __syncthreads();

#pragma unroll
    for (int mt = 0; mt < 2; ++mt) {
#pragma unroll
        for (int nt = 0; nt < 8; ++nt) {
            float bn = bias[n0 + nt * 16 + r];
#pragma unroll
            for (int e = 0; e < 8; ++e) acc[mt][nt][e] += bn;
        }
#pragma unroll
        for (int e = 0; e < 8; ++e) {
            float s = 0.0f;
#pragma unroll
            for (int nt = 0; nt < 8; ++nt) { float c = acc[mt][nt][e]; s += c * c; }
            atomicAdd(&rowsq[mt * 16 + half * 8 + e], s);
        }
    }
    __syncthreads();

#pragma unroll
    for (int mt = 0; mt < 2; ++mt)
#pragma unroll
        for (int e = 0; e < 8; ++e) {
            int row = mt * 16 + half * 8 + e;
            float scale = 1.0f / fmaxf(sqrtf(rowsq[row]), 1e-12f);
            size_t base = (size_t)(m0 + row) * HDIM;
#pragma unroll
            for (int nt = 0; nt < 8; ++nt) {
                int gn = n0 + nt * 16 + r;
                out16[base + gn] = (_Float16)(acc[mt][nt][e] * scale);
            }
        }
}

// ---------------------------------------------------------------------------
// Kernel: out_f32[M,N] = A16[.,1024] * B16[N,1024]^T   (pure f16 WMMA)
// Block 256 thr (8 waves). grid.x = N/128, grid.y = M/16. Wave = 16M x 16N.
// ---------------------------------------------------------------------------
__global__ __launch_bounds__(256) void k_gemm_f16(
    const _Float16* __restrict__ A, const _Float16* __restrict__ B,
    float* __restrict__ out, int N, int K)
{
    const int tid  = threadIdx.x;
    const int wave = tid >> 5;
    const int lane = tid & 31;
    const int half = lane >> 4;
    const int r    = lane & 15;
    const int m0   = blockIdx.y * 16;
    const int n    = blockIdx.x * 128 + wave * 16;

    v8f acc = v8f_zero();
    for (int kk = 0; kk < K; kk += 32) {
        const _Float16* arow = A + (size_t)(m0 + r) * K + kk;
        const _Float16* brow = B + (size_t)(n  + r) * K + kk;
        v16h af = frag_from_f16(arow + half * 8, arow + 16 + half * 8);
        v16h bf = frag_from_f16(brow + half * 8, brow + 16 + half * 8);
        acc = wmma_f16(af, bf, acc);
    }
#pragma unroll
    for (int e = 0; e < 8; ++e) {
        int row = half * 8 + e;
        out[(size_t)(m0 + row) * N + n + r] = acc[e];
    }
}

// ---------------------------------------------------------------------------
// fp32 -> f16 streaming conversion (float4 in, v4h out); n4 = n/4
// ---------------------------------------------------------------------------
__global__ void k_f32_to_f16(const float* __restrict__ in,
                             _Float16* __restrict__ out, int n4)
{
    int i = blockIdx.x * 256 + threadIdx.x;
    if (i >= n4) return;
    float4 v = ((const float4*)in)[i];
    v4h h;
    h[0] = (_Float16)v.x; h[1] = (_Float16)v.y;
    h[2] = (_Float16)v.z; h[3] = (_Float16)v.w;
    ((v4h*)out)[i] = h;
}

// ---------------------------------------------------------------------------
// Builders for concatenated A matrices (fp32, padded rows zeroed)
// ---------------------------------------------------------------------------
__global__ void k_build_Arq(const float* __restrict__ roles,
                            const float* __restrict__ q, float* __restrict__ out)
{
    size_t idx = (size_t)blockIdx.x * 256 + threadIdx.x;   // 32 x 8192
    if (idx >= (size_t)32 * 8192) return;
    int row = (int)(idx / 8192), col = (int)(idx % 8192);
    float v = 0.0f;
    if (row < NROLE) v = (col < D_IN) ? roles[(size_t)row * D_IN + col] : q[col - D_IN];
    out[idx] = v;
}

__global__ void k_build_Arrq(const float* __restrict__ roles,
                             const float* __restrict__ q, float* __restrict__ out)
{
    size_t idx = (size_t)blockIdx.x * 256 + threadIdx.x;   // 64 x 12288
    if (idx >= (size_t)64 * 12288) return;
    int row = (int)(idx / 12288), col = (int)(idx % 12288);
    int i = row >> 3, j = row & 7;
    float v;
    if (col < D_IN)            v = roles[(size_t)i * D_IN + col];
    else if (col < 2 * D_IN)   v = roles[(size_t)j * D_IN + (col - D_IN)];
    else                       v = q[col - 2 * D_IN];
    out[idx] = v;
}

// ---------------------------------------------------------------------------
// loop_score (8) / edge_score (64): sigmoid(dot(row, W) + b)
// grid = 72 blocks x 128 threads
// ---------------------------------------------------------------------------
__global__ void k_scores(const _Float16* __restrict__ rq16,
                         const _Float16* __restrict__ rrq16,
                         const float* __restrict__ W_lp, const float* __restrict__ b_lp,
                         const float* __restrict__ W_ep, const float* __restrict__ b_ep,
                         float* __restrict__ loop_score, float* __restrict__ edge_score)
{
    __shared__ float red[128];
    int b = blockIdx.x, t = threadIdx.x;
    const _Float16* row; const float* w; float bb; float* dst;
    if (b < NROLE) { row = rq16  + (size_t)b * HDIM; w = W_lp; bb = b_lp[0]; dst = &loop_score[b]; }
    else { int e = b - NROLE; row = rrq16 + (size_t)e * HDIM; w = W_ep; bb = b_ep[0]; dst = &edge_score[e]; }
    float s = 0.0f;
    for (int k = t; k < HDIM; k += 128) s += (float)row[k] * w[k];
    red[t] = s; __syncthreads();
    for (int off = 64; off; off >>= 1) { if (t < off) red[t] += red[t + off]; __syncthreads(); }
    if (t == 0) *dst = 1.0f / (1.0f + expf(-(red[0] + bb)));
}

// ---------------------------------------------------------------------------
// Row softmax: one block per row, 256 threads
// ---------------------------------------------------------------------------
__global__ void k_softmax(const float* __restrict__ in, float* __restrict__ out, int len)
{
    __shared__ float red[256];
    int row = blockIdx.x, t = threadIdx.x;
    const float* x = in  + (size_t)row * len;
    float*       y = out + (size_t)row * len;
    float m = -3.4e38f;
    for (int k = t; k < len; k += 256) m = fmaxf(m, x[k]);
    red[t] = m; __syncthreads();
    for (int off = 128; off; off >>= 1) { if (t < off) red[t] = fmaxf(red[t], red[t + off]); __syncthreads(); }
    m = red[0]; __syncthreads();
    float s = 0.0f;
    for (int k = t; k < len; k += 256) { float e = expf(x[k] - m); y[k] = e; s += e; }
    red[t] = s; __syncthreads();
    for (int off = 128; off; off >>= 1) { if (t < off) red[t] += red[t + off]; __syncthreads(); }
    float inv = 1.0f / red[0];
    for (int k = t; k < len; k += 256) y[k] *= inv;
}

// ---------------------------------------------------------------------------
// PCG64 (numpy default_rng) — portable 128-bit step
// ---------------------------------------------------------------------------
__host__ __device__ __forceinline__ void mul64wide(
    unsigned long long a, unsigned long long b,
    unsigned long long& hi, unsigned long long& lo)
{
    unsigned long long a0 = a & 0xffffffffull, a1 = a >> 32;
    unsigned long long b0 = b & 0xffffffffull, b1 = b >> 32;
    unsigned long long p00 = a0 * b0, p01 = a0 * b1, p10 = a1 * b0, p11 = a1 * b1;
    unsigned long long mid = (p00 >> 32) + (p01 & 0xffffffffull) + (p10 & 0xffffffffull);
    lo = (p00 & 0xffffffffull) | (mid << 32);
    hi = p11 + (p01 >> 32) + (p10 >> 32) + (mid >> 32);
}

#define PCG_MHI 0x2360ed051fc65da4ULL
#define PCG_MLO 0x4385df649fccf645ULL

__host__ __device__ __forceinline__ void pcg_step(
    unsigned long long& shi, unsigned long long& slo,
    unsigned long long ihi, unsigned long long ilo)
{
    unsigned long long hi, lo;
    mul64wide(slo, PCG_MLO, hi, lo);
    hi += slo * PCG_MHI + shi * PCG_MLO;
    unsigned long long nlo = lo + ilo;
    unsigned long long c = (nlo < lo) ? 1ull : 0ull;
    shi = hi + ihi + c;
    slo = nlo;
}

__device__ __forceinline__ unsigned long long pcg_next64(
    unsigned long long& shi, unsigned long long& slo,
    unsigned long long ihi, unsigned long long ilo)
{
    pcg_step(shi, slo, ihi, ilo);
    unsigned long long x = shi ^ slo;
    unsigned int rot = (unsigned int)(shi >> 58);   // state >> 122
    return rot ? ((x >> rot) | (x << (64u - rot))) : x;
}

__device__ __forceinline__ double pcg_double(
    unsigned long long& shi, unsigned long long& slo,
    unsigned long long ihi, unsigned long long ilo)
{
    return (double)(pcg_next64(shi, slo, ihi, ilo) >> 11) * (1.0 / 9007199254740992.0);
}

__device__ int choice_cdf(const float* __restrict__ p, int n, double u)
{
    double s = 0.0;
    for (int k = 0; k < n; ++k) s += (double)p[k];
    double cum = 0.0;
    for (int k = 0; k < n; ++k) { cum += (double)p[k] / s; if (u < cum) return k; }
    return n - 1;
}

// ---------------------------------------------------------------------------
// Sequential sampler (replicates the reference Python loop). One thread.
// d_out = [edge0.i, edge0.j, edge0.k, edge1..., log_probs]  as floats.
// ---------------------------------------------------------------------------
__global__ void k_sample(const float* __restrict__ ls, const float* __restrict__ lp,
                         const float* __restrict__ es, const float* __restrict__ ep,
                         float* __restrict__ out, int out_size,
                         unsigned long long shi, unsigned long long slo,
                         unsigned long long ihi, unsigned long long ilo)
{
    if (threadIdx.x != 0 || blockIdx.x != 0) return;
    float logp = 0.0f;
    int ne = 0;
    unsigned int adj[NROLE];
    for (int i = 0; i < NROLE; ++i) adj[i] = 0u;

    // Phase 1: self-loops
    for (int i = 0; i < NROLE; ++i) {
        double rr = pcg_double(shi, slo, ihi, ilo);
        float sc = ls[i];
        if (rr < (double)sc) {
            double u = pcg_double(shi, slo, ihi, ilo);
            int k = choice_cdf(lp + (size_t)i * NLOOP, NLOOP, u);
            if (3 * ne + 2 < out_size) {
                out[3 * ne + 0] = (float)i;
                out[3 * ne + 1] = (float)i;
                out[3 * ne + 2] = (float)k;
            }
            ++ne;
            logp += logf(sc) + logf(lp[(size_t)i * NLOOP + k] + EPS_F);
        } else {
            logp += logf(1.0f - sc + EPS_F);
        }
    }

    // Phase 2: directed edges with DAG constraint
    for (int i = 0; i < NROLE; ++i) {
        for (int j = 0; j < NROLE; ++j) {
            if (i == j) continue;
            double rr = pcg_double(shi, slo, ihi, ilo);
            float sc = es[i * NROLE + j];
            if (rr < (double)sc) {
                double u = pcg_double(shi, slo, ihi, ilo);
                int k = choice_cdf(ep + (size_t)(i * NROLE + j) * NEDGE, NEDGE, u);
                // cycle iff j already reaches i
                unsigned int reach = 1u << j;
                for (int it = 0; it < NROLE; ++it) {
                    unsigned int nr = reach;
                    for (int v = 0; v < NROLE; ++v)
                        if ((reach >> v) & 1u) nr |= adj[v];
                    reach = nr;
                }
                if (!((reach >> i) & 1u)) {
                    adj[i] |= 1u << j;
                    if (3 * ne + 2 < out_size) {
                        out[3 * ne + 0] = (float)i;
                        out[3 * ne + 1] = (float)j;
                        out[3 * ne + 2] = (float)k;
                    }
                    ++ne;
                    logp += logf(sc) + logf(ep[(size_t)(i * NROLE + j) * NEDGE + k] + EPS_F);
                }
            } else {
                logp += logf(1.0f - sc + EPS_F);
            }
        }
    }
    if (out_size > 0) out[out_size - 1] = logp;
}

// ---------------------------------------------------------------------------
// Host-side numpy SeedSequence(0) -> PCG64 state (pure CPU, capture-safe)
// ---------------------------------------------------------------------------
static void np_pcg64_from_seed0(unsigned long long& shi, unsigned long long& slo,
                                unsigned long long& ihi, unsigned long long& ilo)
{
    const uint32_t XSHIFT = 16u;
    const uint32_t INIT_A = 0x43b0d7e5u, MULT_A = 0x931e8875u;
    const uint32_t INIT_B = 0x8b51f9ddu, MULT_B = 0x58f38dedu;
    const uint32_t MIX_L  = 0xca01f9ddu, MIX_R  = 0x4973f715u;

    uint32_t pool[4];
    uint32_t hc = INIT_A;
    auto hashmix = [&](uint32_t v) -> uint32_t {
        v ^= hc; hc *= MULT_A; v *= hc; v ^= v >> XSHIFT; return v;
    };
    auto mix = [&](uint32_t x, uint32_t y) -> uint32_t {
        uint32_t r = x * MIX_L - y * MIX_R; r ^= r >> XSHIFT; return r;
    };
    // entropy = [0]
    for (int i = 0; i < 4; ++i) pool[i] = hashmix(0u);
    for (int is = 0; is < 4; ++is)
        for (int id = 0; id < 4; ++id)
            if (is != id) pool[id] = mix(pool[id], hashmix(pool[is]));

    // generate_state(4, uint64) -> 8 u32 words, little-endian pairing
    uint32_t st[8];
    hc = INIT_B;
    int src = 0;
    for (int i = 0; i < 8; ++i) {
        uint32_t dv = pool[src];
        dv ^= hc; hc *= MULT_B; dv *= hc; dv ^= dv >> XSHIFT;
        st[i] = dv; src = (src + 1) & 3;
    }
    unsigned long long w[4];
    for (int i = 0; i < 4; ++i)
        w[i] = (unsigned long long)st[2 * i] | ((unsigned long long)st[2 * i + 1] << 32);

    // PCG64: initstate = (w0<<64)|w1, initseq = (w2<<64)|w3
    unsigned long long is_hi = w[0], is_lo = w[1];
    unsigned long long iq_hi = w[2], iq_lo = w[3];
    ihi = (iq_hi << 1) | (iq_lo >> 63);
    ilo = (iq_lo << 1) | 1ull;
    shi = 0ull; slo = 0ull;
    pcg_step(shi, slo, ihi, ilo);
    unsigned long long nlo = slo + is_lo;
    shi += is_hi + ((nlo < slo) ? 1ull : 0ull);
    slo = nlo;
    pcg_step(shi, slo, ihi, ilo);
}

// ---------------------------------------------------------------------------
// kernel_launch
// ---------------------------------------------------------------------------
extern "C" void kernel_launch(void* const* d_in, const int* in_sizes, int n_in,
                              void* d_out, int out_size, void* d_ws, size_t ws_size,
                              hipStream_t stream)
{
    const float* q     = (const float*)d_in[0];
    const float* roles = (const float*)d_in[1];
    const float* edgeE = (const float*)d_in[2];
    const float* loopE = (const float*)d_in[3];
    const float* W_rq  = (const float*)d_in[4];
    const float* b_rq  = (const float*)d_in[5];
    const float* W_rrq = (const float*)d_in[6];
    const float* b_rrq = (const float*)d_in[7];
    const float* W_le  = (const float*)d_in[8];
    const float* b_le  = (const float*)d_in[9];
    const float* W_ee  = (const float*)d_in[10];
    const float* b_ee  = (const float*)d_in[11];
    const float* W_lp  = (const float*)d_in[12];
    const float* b_lp  = (const float*)d_in[13];
    const float* W_ep  = (const float*)d_in[14];
    const float* b_ep  = (const float*)d_in[15];

    char* ws = (char*)d_ws;
    auto alloc = [&](size_t bytes) -> char* {
        char* p = ws; ws += (bytes + 255) & ~(size_t)255; return p;
    };
    _Float16* E16    = (_Float16*)alloc((size_t)4096 * 1024 * 2);
    _Float16* L16    = (_Float16*)alloc((size_t)1024 * 1024 * 2);
    _Float16* rq16   = (_Float16*)alloc((size_t)32   * 1024 * 2);
    _Float16* rrq16  = (_Float16*)alloc((size_t)64   * 1024 * 2);
    _Float16* Wee16  = (_Float16*)alloc((size_t)1024 * 4096 * 2);
    _Float16* Wle16  = (_Float16*)alloc((size_t)1024 * 4096 * 2);
    float*    Arq    = (float*)alloc((size_t)32 * 8192  * 4);
    float*    Arrq   = (float*)alloc((size_t)64 * 12288 * 4);
    float*    loopLg = (float*)alloc((size_t)16 * 1024 * 4);
    float*    edgeLg = (float*)alloc((size_t)64 * 4096 * 4);
    float*    loopP  = (float*)alloc((size_t)8  * 1024 * 4);
    float*    edgeP  = (float*)alloc((size_t)64 * 4096 * 4);
    float*    lScore = (float*)alloc(64 * 4);
    float*    eScore = (float*)alloc(64 * 4);

    // Pre-convert heavily reused weights to f16 (L2-resident afterwards)
    {
        int n4 = (1024 * 4096) / 4;
        k_f32_to_f16<<<(n4 + 255) / 256, 256, 0, stream>>>(W_ee, Wee16, n4);
        k_f32_to_f16<<<(n4 + 255) / 256, 256, 0, stream>>>(W_le, Wle16, n4);
    }

    // Build concatenated A matrices (fp32, zero-padded rows)
    k_build_Arq <<<((32 * 8192  + 255) / 256), 256, 0, stream>>>(roles, q, Arq);
    k_build_Arrq<<<((64 * 12288 + 255) / 256), 256, 0, stream>>>(roles, q, Arrq);

    // Fused GEMM + bias + row-L2norm (f16 WMMA, f32 accumulate)
    k_gemm_norm<true ><<<4096 / 32, 256, 0, stream>>>(edgeE, Wee16, b_ee,  E16,   4096);
    k_gemm_norm<true ><<<1024 / 32, 256, 0, stream>>>(loopE, Wle16, b_le,  L16,   4096);
    k_gemm_norm<false><<<1,         256, 0, stream>>>(Arq,   W_rq,  b_rq,  rq16,  8192);
    k_gemm_norm<false><<<2,         256, 0, stream>>>(Arrq,  W_rrq, b_rrq, rrq16, 12288);

    // Logits GEMMs (pure f16 WMMA)
    k_gemm_f16<<<dim3(1024 / 128, 1), 256, 0, stream>>>(rq16,  L16, loopLg, 1024, 1024);
    k_gemm_f16<<<dim3(4096 / 128, 4), 256, 0, stream>>>(rrq16, E16, edgeLg, 4096, 1024);

    // Sigmoid scores + softmaxes
    k_scores <<<72, 128, 0, stream>>>(rq16, rrq16, W_lp, b_lp, W_ep, b_ep, lScore, eScore);
    k_softmax<<<8,  256, 0, stream>>>(loopLg, loopP, 1024);
    k_softmax<<<64, 256, 0, stream>>>(edgeLg, edgeP, 4096);

    // Sequential PCG64 sampling (numpy default_rng(0) replicated)
    unsigned long long shi, slo, ihi, ilo;
    np_pcg64_from_seed0(shi, slo, ihi, ilo);
    k_sample<<<1, 1, 0, stream>>>(lScore, loopP, eScore, edgeP,
                                  (float*)d_out, out_size, shi, slo, ihi, ilo);
}
